// HEP_78881369358777
// MI455X (gfx1250) — matrix operand order repaired and checked by reading, not compiled, not run on hardware
//
#include <hip/hip_runtime.h>
#include <hip/hip_bf16.h>

#define B_ 128
#define T_ 128
#define D_ 128
#define H_ 256
#define S_ 32
#define E_ 768
#define FIN 512          /* D + H + D */
#define AH  384          /* H + D     */
#define ASTR 520         /* f16 LDS row stride (16B-aligned, padded) */

typedef __attribute__((ext_vector_type(16))) _Float16 v16h;
typedef __attribute__((ext_vector_type(8)))  _Float16 v8h;
typedef __attribute__((ext_vector_type(8)))  float    v8f;

union F16Frag { v16h v; v8h h[2]; };

// A-fragment (16x32 f16) from LDS. Dense A layout: lane holds K = base..base+7
// (elems 0..7) and 16+base..16+base+7 (elems 8..15), base = (lane>=16)?8:0.
__device__ inline v16h ldsA(const _Float16* base, int lane, int kt) {
  F16Frag u;
  const _Float16* p = base + (lane & 15) * ASTR + kt * 32 + ((lane >> 4) << 3);
  u.h[0] = *(const v8h*)p;
  u.h[1] = *(const v8h*)(p + 16);
  return u.v;
}

// B-fragment (32x16 f16, column n = lane&15) from global (N,K) row-major f16
// weights: lane holds K = base..base+7 (elems 0..7) and 16+base..16+base+7
// (elems 8..15), base = (lane>=16)?8:0.
__device__ inline v16h gblB(const _Float16* W, int K, int n0, int lane, int kt) {
  F16Frag u;
  const _Float16* p = W + (size_t)(n0 + (lane & 15)) * K + kt * 32 + ((lane >> 4) << 3);
  u.h[0] = *(const v8h*)p;
  u.h[1] = *(const v8h*)(p + 16);
  return u.v;
}

__device__ inline float sigmoidf_(float x) { return 1.0f / (1.0f + __expf(-x)); }

__global__ void convert_kernel(const float* __restrict__ src, _Float16* __restrict__ dst, int n) {
  int i = blockIdx.x * blockDim.x + threadIdx.x;
  if (i < n) dst[i] = (_Float16)src[i];
}

// Persistent bidirectional GRU-D scan. grid = 16 (8 batch tiles x 2 dirs), block = 128.
__global__ __launch_bounds__(128)
void hep_scan_kernel(const float* __restrict__ x_long,
                     const float* __restrict__ x_mean_g,
                     const float* __restrict__ zl_b, const float* __restrict__ rl_b,
                     const float* __restrict__ hl_b,
                     const float* __restrict__ gx_w, const float* __restrict__ gx_b,
                     const float* __restrict__ gh_b,
                     const _Float16* __restrict__ wz, const _Float16* __restrict__ wr,
                     const _Float16* __restrict__ wh, const _Float16* __restrict__ wgh,
                     const _Float16* __restrict__ watt,
                     float* __restrict__ gHsum, float* __restrict__ gHsq,
                     float* __restrict__ gAtt /* [2][B][D] */)
{
  const int tid  = threadIdx.x;
  const int lane = tid & 31;
  const int w    = tid >> 5;
  const int tile = blockIdx.x & 7;
  const int dir  = blockIdx.x >> 3;
  const int b0   = tile * 16;

  __shared__ __align__(32) float    sh_h[16][H_];     // hidden state
  __shared__ __align__(32) float    sh_xm[16][D_];    // x_mean
  __shared__ __align__(32) float    sh_x[16][D_];     // current x
  __shared__ __align__(32) float    sh_att[16][D_];   // attention logits
  __shared__ __align__(32) _Float16 sh_A[16][ASTR];   // WMMA A-matrix buffer
  __shared__ float sh_red[16][8];

  // ---- Prologue: x_mean per (b,d); zero h. thread = column d, loop rows.
  for (int r = 0; r < 16; ++r) {
    const int b = b0 + r;
    const float* px = x_long + ((size_t)b * 7 + 0) * (T_ * D_) + tid;  // x_seq
    const float* pm = x_long + ((size_t)b * 7 + 3) * (T_ * D_) + tid;  // var_mask
    float sx = 0.f, cnt = 0.f;
    for (int t = 0; t < T_; ++t) { sx += px[(size_t)t * D_]; cnt += pm[(size_t)t * D_]; }
    float cur = sx / fmaxf(cnt, 1.0f);
    float lam = tanhf(cnt);
    sh_xm[r][tid] = x_mean_g[tid] * (1.f - lam) + lam * cur;
    sh_h[r][tid]        = 0.f;
    sh_h[r][tid + 128]  = 0.f;
  }

  // per-lane biases for owned output columns (c = 64w + 16j + lane%16)
  float zlb[4], rlb[4], hlb[4], ghb[4];
  for (int j = 0; j < 4; ++j) {
    int c = 64 * w + 16 * j + (lane & 15);
    zlb[j] = zl_b[c]; rlb[j] = rl_b[c]; hlb[j] = hl_b[c]; ghb[j] = gh_b[c];
  }
  const float gxw = gx_w[tid];
  const float gxb = gx_b[tid];

  float hs_acc[4][8], hq_acc[4][8];
  for (int j = 0; j < 4; ++j) for (int v = 0; v < 8; ++v) { hs_acc[j][v] = 0.f; hq_acc[j][v] = 0.f; }
  float aacc[16];
  for (int i = 0; i < 16; ++i) aacc[i] = 0.f;

  const int c_xl = dir ? 2 : 1;   // x_last vs x_last_rev
  const int c_dl = dir ? 5 : 4;   // delta vs delta_rev
  const int half = (lane >> 4) << 3;

  __syncthreads();

#pragma unroll 1
  for (int step = 0; step < T_; ++step) {
    const int t = dir ? (T_ - 1 - step) : step;

    // Launder a zero offset (not the pointers!) each step:
    //  - defeats LICM's hoist of the read-only B-fragment loads (which would
    //    otherwise preload ~1MB of fragments and spill them to scratch), and
    //  - keeps pointer provenance intact so InferAddressSpaces still proves
    //    the bases global -> global_load_b128 (LOADcnt only), not
    //    flat_load_b128 (which ties up DScnt too and entangles the weight
    //    stream with the LDS A-fragment waits).
    int lzero = 0;
    asm volatile("" : "+s"(lzero));
    const _Float16* wz_s   = wz   + lzero;
    const _Float16* wr_s   = wr   + lzero;
    const _Float16* wh_s   = wh   + lzero;
    const _Float16* wgh_s  = wgh  + lzero;
    const _Float16* watt_s = watt + lzero;

    // Phase 1: load inputs, GRU-D imputation; stash dl (A cols 0:128) + mask (384:512)
    for (int r = 0; r < 16; ++r) {
      const int b = b0 + r;
      const size_t bo = (size_t)b * 7 * T_ * D_ + (size_t)t * D_ + tid;
      float xv = x_long[bo];
      float xl = x_long[bo + (size_t)c_xl * T_ * D_];
      float mk = x_long[bo + 3 * (size_t)T_ * D_];
      float dl = x_long[bo + (size_t)c_dl * T_ * D_];
      float dx = __expf(fmaxf(dl * gxw + gxb, 0.f));
      float nx = mk * xv + (1.f - mk) * (dx * xl + (1.f - dx) * sh_xm[r][tid]);
      sh_x[r][tid]        = nx;
      sh_A[r][384 + tid]  = (_Float16)mk;
      sh_A[r][tid]        = (_Float16)dl;
    }
    __syncthreads();

    // Phase 2: delta_h pre-activation = dl @ gh_w^T   (M16 N256 K128)
    v8f dh[4] = {};
    for (int kt = 0; kt < 4; ++kt) {
      v16h a = ldsA(&sh_A[0][0], lane, kt);
      for (int j = 0; j < 4; ++j) {
        v16h bb = gblB(wgh_s, D_, 64 * w + 16 * j, lane, kt);
        dh[j] = __builtin_amdgcn_wmma_f32_16x16x32_f16(false, a, false, bb, (short)0, dh[j], false, false);
      }
    }
    __syncthreads();

    // Phase 3: A := tanh([x, h_old])  (cols 0:384)
    for (int r = 0; r < 16; ++r) sh_A[r][tid] = (_Float16)tanhf(sh_x[r][tid]);
    for (int k = 0; k < 32; ++k) {
      int pos = tid + 128 * k; int r = pos >> 8; int c = pos & 255;
      sh_A[r][128 + c] = (_Float16)tanhf(sh_h[r][c]);
    }
    __syncthreads();

    // Phase 4: attention logits = a @ att_w^T (M16 N128 K384); then h *= delta_h
    {
      v8f at[2] = {};
      for (int kt = 0; kt < 12; ++kt) {
        v16h a = ldsA(&sh_A[0][0], lane, kt);
        for (int j = 0; j < 2; ++j) {
          v16h bb = gblB(watt_s, AH, 32 * w + 16 * j, lane, kt);
          at[j] = __builtin_amdgcn_wmma_f32_16x16x32_f16(false, a, false, bb, (short)0, at[j], false, false);
        }
      }
      for (int j = 0; j < 2; ++j)
        for (int v = 0; v < 8; ++v)
          sh_att[v + half][32 * w + 16 * j + (lane & 15)] = at[j][v];
      // decay owned h columns (a-matrix already captured old h)
      for (int j = 0; j < 4; ++j)
        for (int v = 0; v < 8; ++v) {
          float d = __expf(fmaxf(dh[j][v] + ghb[j], 0.f));
          sh_h[v + half][64 * w + 16 * j + (lane & 15)] *= d;
        }
    }
    __syncthreads();

    // Phase 5: softmax over D, x := att*x, accumulate att*seq_mask; write comb x (f16)
    {
      int r = tid >> 3, g = tid & 7, cb = g * 16;
      float m = -1e30f;
      for (int i = 0; i < 16; ++i) m = fmaxf(m, sh_att[r][cb + i]);
      sh_red[r][g] = m;
      __syncthreads();
      float rm = sh_red[r][0];
      for (int i = 1; i < 8; ++i) rm = fmaxf(rm, sh_red[r][i]);
      float e[16]; float s = 0.f;
      for (int i = 0; i < 16; ++i) { e[i] = __expf(sh_att[r][cb + i] - rm); s += e[i]; }
      __syncthreads();
      sh_red[r][g] = s;
      __syncthreads();
      float rs = 0.f;
      for (int i = 0; i < 8; ++i) rs += sh_red[r][i];
      float inv = 1.f / rs;
      const int b = b0 + r;
      const float* smp = x_long + ((size_t)b * 7 + 6) * (T_ * D_) + (size_t)t * D_ + cb;
      for (int i = 0; i < 16; ++i) {
        float att = e[i] * inv;
        float nx  = att * sh_x[r][cb + i];
        sh_x[r][cb + i] = nx;
        sh_A[r][cb + i] = (_Float16)nx;
        aacc[i] += att * smp[i];
      }
    }
    __syncthreads();

    // Phase 6: comb h part (cols 128:384) = decayed h
    for (int k = 0; k < 32; ++k) {
      int pos = tid + 128 * k; int r = pos >> 8; int c = pos & 255;
      sh_A[r][128 + c] = (_Float16)sh_h[r][c];
    }
    __syncthreads();

    // Phase 7: z & r GEMMs  (M16 N256 K512 each)
    v8f zac[4] = {}, rac[4] = {};
    for (int kt = 0; kt < 16; ++kt) {
      v16h a = ldsA(&sh_A[0][0], lane, kt);
      for (int j = 0; j < 4; ++j) {
        v16h bz = gblB(wz_s, FIN, 64 * w + 16 * j, lane, kt);
        zac[j] = __builtin_amdgcn_wmma_f32_16x16x32_f16(false, a, false, bz, (short)0, zac[j], false, false);
        v16h br = gblB(wr_s, FIN, 64 * w + 16 * j, lane, kt);
        rac[j] = __builtin_amdgcn_wmma_f32_16x16x32_f16(false, a, false, br, (short)0, rac[j], false, false);
      }
    }
    __syncthreads();

    // Phase 8: comb_r middle = sigmoid(r)*h  (own columns)
    for (int j = 0; j < 4; ++j)
      for (int v = 0; v < 8; ++v) {
        int r = v + half, c = 64 * w + 16 * j + (lane & 15);
        float rr = sigmoidf_(rac[j][v] + rlb[j]);
        sh_A[r][128 + c] = (_Float16)(rr * sh_h[r][c]);
      }
    __syncthreads();

    // Phase 9: h_tilde GEMM  (M16 N256 K512)
    v8f hac[4] = {};
    for (int kt = 0; kt < 16; ++kt) {
      v16h a = ldsA(&sh_A[0][0], lane, kt);
      for (int j = 0; j < 4; ++j) {
        v16h bh = gblB(wh_s, FIN, 64 * w + 16 * j, lane, kt);
        hac[j] = __builtin_amdgcn_wmma_f32_16x16x32_f16(false, a, false, bh, (short)0, hac[j], false, false);
      }
    }
    // Phase 10: h = (1-z)h + z*tanh(ht); accumulate sum / sum^2 for BN+pool
    for (int j = 0; j < 4; ++j)
      for (int v = 0; v < 8; ++v) {
        int r = v + half, c = 64 * w + 16 * j + (lane & 15);
        float z  = sigmoidf_(zac[j][v] + zlb[j]);
        float ht = tanhf(hac[j][v] + hlb[j]);
        float hn = (1.f - z) * sh_h[r][c] + z * ht;
        sh_h[r][c] = hn;
        hs_acc[j][v] += hn;
        hq_acc[j][v] += hn * hn;
      }
    __syncthreads();
  }

  // ---- Epilogue: write time-sums (channel offset by direction) and att sums
  for (int j = 0; j < 4; ++j)
    for (int v = 0; v < 8; ++v) {
      int r = v + half, c = 64 * w + 16 * j + (lane & 15);
      size_t idx = (size_t)(b0 + r) * 512 + dir * H_ + c;
      gHsum[idx] = hs_acc[j][v];
      gHsq[idx]  = hq_acc[j][v];
    }
  {
    int r = tid >> 3, cb = (tid & 7) * 16, b = b0 + r;
    for (int i = 0; i < 16; ++i)
      gAtt[(size_t)dir * B_ * D_ + (size_t)b * D_ + cb + i] = aacc[i];
  }
}

__global__ void stats_kernel(const float* __restrict__ gHsum, const float* __restrict__ gHsq,
                             float* __restrict__ mu, float* __restrict__ rstd) {
  int c = blockIdx.x * blockDim.x + threadIdx.x;
  if (c >= 512) return;
  float s = 0.f, q = 0.f;
  for (int b = 0; b < B_; ++b) { s += gHsum[(size_t)b * 512 + c]; q += gHsq[(size_t)b * 512 + c]; }
  const float invN = 1.f / (float)(B_ * T_);
  float m = s * invN;
  float v = q * invN - m * m;
  mu[c]   = m;
  rstd[c] = rsqrtf(v + 1e-5f);
}

__global__ void proj_kernel(const float* __restrict__ out2_w,
                            const float* __restrict__ st_w, const float* __restrict__ st_b,
                            const float* __restrict__ tx_w, const float* __restrict__ tx_b,
                            float* __restrict__ vs, float* __restrict__ vt, float* __restrict__ cterm) {
  int g = blockIdx.x * blockDim.x + threadIdx.x;
  if (g < S_) {
    float s = 0.f;
    for (int h = 0; h < H_; ++h) s += out2_w[512 + h] * st_w[(size_t)h * S_ + g];
    vs[g] = s;
  } else if (g < S_ + E_) {
    int k = g - S_;
    float s = 0.f;
    for (int h = 0; h < H_; ++h) s += out2_w[768 + h] * tx_w[(size_t)h * E_ + k];
    vt[k] = s;
  } else if (g == S_ + E_) {
    float s = 0.f;
    for (int h = 0; h < H_; ++h) s += out2_w[512 + h] * st_b[h] + out2_w[768 + h] * tx_b[h];
    *cterm = s;
  }
}

__global__ __launch_bounds__(128)
void final_kernel(const float* __restrict__ gHsum,
                  const float* __restrict__ mu, const float* __restrict__ rstd,
                  const float* __restrict__ bn_g, const float* __restrict__ bn_b,
                  const float* __restrict__ out2_w, const float* __restrict__ out2_b,
                  const float* __restrict__ x_cs, const float* __restrict__ x_text,
                  const float* __restrict__ vs, const float* __restrict__ vt,
                  const float* __restrict__ cterm,
                  const float* __restrict__ gAtt, const float* __restrict__ x_long,
                  float* __restrict__ out) {
  const int b = blockIdx.x, tid = threadIdx.x;
  __shared__ float red[128];
  float p = 0.f;
  for (int c = tid; c < 512; c += 128) {
    float hp  = gHsum[(size_t)b * 512 + c] * (1.f / (float)T_);
    float val = (hp - mu[c]) * rstd[c] * bn_g[c] + bn_b[c];
    p += val * out2_w[c];
  }
  if (tid < S_) p += x_cs[(size_t)b * S_ + tid] * vs[tid];
  for (int k = tid; k < E_; k += 128) p += x_text[(size_t)b * E_ + k] * vt[k];
  red[tid] = p;
  __syncthreads();
  for (int s = 64; s > 0; s >>= 1) { if (tid < s) red[tid] += red[tid + s]; __syncthreads(); }
  if (tid == 0) out[b] = red[0] + *cterm + out2_b[0];
  // mean_att: d = tid
  const float* smp = x_long + ((size_t)b * 7 + 6) * (T_ * D_) + tid;
  float ssum = 0.f;
  for (int tt = 0; tt < T_; ++tt) ssum += smp[(size_t)tt * D_];
  float a = gAtt[(size_t)b * D_ + tid] + gAtt[(size_t)B_ * D_ + (size_t)b * D_ + tid];
  out[B_ + (size_t)b * D_ + tid] = a / (2.f * ssum);
}

extern "C" void kernel_launch(void* const* d_in, const int* in_sizes, int n_in,
                              void* d_out, int out_size, void* d_ws, size_t ws_size,
                              hipStream_t stream) {
  (void)in_sizes; (void)n_in; (void)out_size; (void)ws_size;
  const float* x_long   = (const float*)d_in[0];
  const float* x_cs     = (const float*)d_in[1];
  const float* x_text   = (const float*)d_in[2];
  const float* x_mean_g = (const float*)d_in[3];
  const float* zl_w     = (const float*)d_in[4];
  const float* zl_b     = (const float*)d_in[5];
  const float* rl_w     = (const float*)d_in[6];
  const float* rl_b     = (const float*)d_in[7];
  const float* hl_w     = (const float*)d_in[8];
  const float* hl_b     = (const float*)d_in[9];
  const float* gx_w     = (const float*)d_in[10];
  const float* gx_b     = (const float*)d_in[11];
  const float* gh_w     = (const float*)d_in[12];
  const float* gh_b     = (const float*)d_in[13];
  const float* att_w    = (const float*)d_in[14];
  const float* st_w     = (const float*)d_in[15];
  const float* st_b     = (const float*)d_in[16];
  const float* tx_w     = (const float*)d_in[17];
  const float* tx_b     = (const float*)d_in[18];
  const float* out2_w   = (const float*)d_in[19];
  const float* out2_b   = (const float*)d_in[20];
  const float* bn_g     = (const float*)d_in[21];
  const float* bn_b     = (const float*)d_in[22];

  char* ws = (char*)d_ws;
  _Float16* wz   = (_Float16*)(ws + 0);        // 256*512 f16
  _Float16* wr   = (_Float16*)(ws + 262144);
  _Float16* wh   = (_Float16*)(ws + 524288);
  _Float16* wgh  = (_Float16*)(ws + 786432);   // 256*128 f16
  _Float16* watt = (_Float16*)(ws + 851968);   // 128*384 f16
  float* hsum  = (float*)(ws + 950272);        // [B][512]
  float* hsq   = (float*)(ws + 1212416);       // [B][512]
  float* attb  = (float*)(ws + 1474560);       // [2][B][D]
  float* mu    = (float*)(ws + 1605632);       // [512]
  float* rstd  = (float*)(ws + 1607680);       // [512]
  float* vs    = (float*)(ws + 1609728);       // [32]
  float* vt    = (float*)(ws + 1609856);       // [768]
  float* cterm = (float*)(ws + 1612928);       // [1]

  int n;
  n = H_ * FIN; convert_kernel<<<(n + 255) / 256, 256, 0, stream>>>(zl_w, wz, n);
  n = H_ * FIN; convert_kernel<<<(n + 255) / 256, 256, 0, stream>>>(rl_w, wr, n);
  n = H_ * FIN; convert_kernel<<<(n + 255) / 256, 256, 0, stream>>>(hl_w, wh, n);
  n = H_ * D_;  convert_kernel<<<(n + 255) / 256, 256, 0, stream>>>(gh_w, wgh, n);
  n = D_ * AH;  convert_kernel<<<(n + 255) / 256, 256, 0, stream>>>(att_w, watt, n);

  hep_scan_kernel<<<16, 128, 0, stream>>>(x_long, x_mean_g, zl_b, rl_b, hl_b,
                                          gx_w, gx_b, gh_b,
                                          wz, wr, wh, wgh, watt,
                                          hsum, hsq, attb);

  stats_kernel<<<2, 256, 0, stream>>>(hsum, hsq, mu, rstd);
  proj_kernel<<<4, 256, 0, stream>>>(out2_w, st_w, st_b, tx_w, tx_b, vs, vt, cterm);
  final_kernel<<<128, 128, 0, stream>>>(hsum, mu, rstd, bn_g, bn_b, out2_w, out2_b,
                                        x_cs, x_text, vs, vt, cterm,
                                        attb, x_long, (float*)d_out);
}
